// MambaBlock_48069273977399
// MI455X (gfx1250) — matrix operand order repaired and checked by reading, not compiled, not run on hardware
//
#include <hip/hip_runtime.h>
#include <hip/hip_bf16.h>

#define D_MODEL   1024
#define D_INNER   2048
#define HEADDIM   64
#define NHEADS    32
#define D_STATE   128
#define D_CONV    4
#define CONV_DIM  (D_INNER + 2 * D_STATE)              // 2304
#define D_IN_PROJ (2 * D_INNER + 2 * D_STATE + NHEADS) // 4384
#define EPS_F     1e-05f
#define BATCH     2
#define SEQLEN    2048
#define ROWS      (BATCH * SEQLEN)                     // 4096

typedef __attribute__((ext_vector_type(16))) __bf16 v16bf;
typedef __attribute__((ext_vector_type(8)))  float  v8f;

// ---------------------------------------------------------------------------
// gfx1250 async global->LDS path (guarded; falls back to sync LDS stores).
// Confirmed on this toolchain: 4 args, generic pointers to int32x4.
// ---------------------------------------------------------------------------
#if __has_builtin(__builtin_amdgcn_global_load_async_to_lds_b128) && \
    __has_builtin(__builtin_amdgcn_s_wait_asynccnt)
#define HAVE_ASYNC_LDS 1
#else
#define HAVE_ASYNC_LDS 0
#endif

#if HAVE_ASYNC_LDS
typedef int av4i __attribute__((vector_size(16)));
__device__ __forceinline__ void async_copy_b128(const void* g, void* l) {
    __builtin_amdgcn_global_load_async_to_lds_b128(
        (av4i*)g, (av4i*)l, 0, 0);
}
#endif

// ---------------------------------------------------------------------------
// gfx1250 Tensor Data Mover path (guarded). D# per CDNA5 ISA ch.8:
//  group0: [1:0]=count, [63:32]=lds_addr, [120:64]=global_addr, [127:126]=2
//  group1: [17:16]=data_size, [20]=pad_enable, [24:22]=pad_interval,
//          [31:25]=pad_amount, [79:48]=tensor_dim0, [111:80]=tensor_dim1,
//          [127:112]=tile_dim0, [143:128]=tile_dim1, [207:160]=dim0_stride
// ---------------------------------------------------------------------------
#if __has_builtin(__builtin_amdgcn_tensor_load_to_lds) && \
    __has_builtin(__builtin_amdgcn_s_wait_tensorcnt)
#define HAVE_TDM 1
#else
#define HAVE_TDM 0
#endif

#if HAVE_TDM
typedef unsigned int tdmv4u __attribute__((vector_size(16)));
typedef int          tdmv4i __attribute__((vector_size(16)));
typedef int          tdmv8i __attribute__((vector_size(32)));

// Load a 2D tile (tile_n rows x tile_k bf16 elems, row stride K_ elems) from
// global to LDS, inserting pad_amount(+1) DWORDs after every
// 2^(pad_interval+1) DWORDs stored.
__device__ __forceinline__ void tdm_load_tile_2d(
    const void* gsrc, unsigned lds_off, int K_, int tile_k, int tile_n,
    unsigned pad_interval_code, unsigned pad_amount_code)
{
    unsigned long long ga = (unsigned long long)(uintptr_t)gsrc;
    tdmv4u g0 = { 1u,                                   // count=1 (user D#)
                  lds_off,                              // lds_addr (bytes)
                  (unsigned)(ga & 0xFFFFFFFFu),         // global_addr[31:0]
                  (unsigned)((ga >> 32) & 0x1FFFFFFu) | (2u << 30) }; // +type=2
    tdmv8i g1 = {
        (int)((1u << 16) | (1u << 20) |                 // data_size=2B, pad_en
              (pad_interval_code << 22) | (pad_amount_code << 25)),
        (int)(((unsigned)K_ & 0xFFFFu) << 16),          // tensor_dim0[15:0]
        (int)(((unsigned)K_ >> 16) |                    // tensor_dim0[31:16]
              (((unsigned)tile_n & 0xFFFFu) << 16)),    // tensor_dim1[15:0]
        (int)(((unsigned)tile_k & 0xFFFFu) << 16),      // tile_dim0
        (int)((unsigned)tile_n & 0xFFFFu),              // tile_dim1 (tile_dim2=0)
        (int)K_,                                        // dim0_stride[31:0]
        0, 0 };
    tdmv4i gz = { 0, 0, 0, 0 };
#if __clang_major__ >= 23
    tdmv8i gz8 = { 0, 0, 0, 0, 0, 0, 0, 0 };
    __builtin_amdgcn_tensor_load_to_lds(g0, g1, gz, gz, gz8, 0);
#else
    __builtin_amdgcn_tensor_load_to_lds(g0, g1, gz, gz, 0);
#endif
}
#endif

__device__ __forceinline__ unsigned short f2bf(float f) {
    unsigned int u = __float_as_uint(f);
    u += 0x7FFFu + ((u >> 16) & 1u);   // round-to-nearest-even
    return (unsigned short)(u >> 16);
}

// ---------------------------------------------------------------------------
// fp32 -> bf16 cast
// ---------------------------------------------------------------------------
__global__ __launch_bounds__(256) void cast_f32_bf16_kernel(
    const float* __restrict__ in, unsigned short* __restrict__ out, long n)
{
    long i = (long)blockIdx.x * blockDim.x + threadIdx.x;
    if (i < n) out[i] = f2bf(in[i]);
}

// ---------------------------------------------------------------------------
// WMMA GEMM: C[M,N] = A[M,K](bf16,row-major) x W[N,K](bf16,row-major)^T
// Block: 8 waves, each wave owns 64 rows x 32 cols (4x2 accumulators).
// Block tile 512x32; B panel (32 x KCHUNK) staged in LDS via TDM (or async,
// or sync fallback) and shared by all 8 waves.
// Requires: M % 512 == 0, N % 32 == 0, K % KCHUNK == 0 (all true here).
// ---------------------------------------------------------------------------
#define KCHUNK  256
#define BSTRIDE (KCHUNK + 8)   // +8 bf16 = +16B = +4 DWORDs pad per row

__global__ __launch_bounds__(256) void wmma_gemm_bf16_kernel(
    const unsigned short* __restrict__ A,
    const unsigned short* __restrict__ W,
    float* __restrict__ C,
    int M, int N, int K)
{
    __shared__ unsigned short Bs[32 * BSTRIDE];

    const int tid  = threadIdx.x;
    const int lane = tid & 31;
    const int wave = tid >> 5;
    const int r    = lane & 15;   // A row / B col / C col within tile
    const int hi   = lane >> 4;   // lane-half selector
    const int kbA  = hi * 8;      // A k-offset (two 8-elem chunks, +16 apart)
    const int kbB  = hi * 8;      // B k-offset (two 8-elem chunks, +16 apart)

    const int m0 = blockIdx.y * 512 + wave * 64;
    const int n0 = blockIdx.x * 32;

    v8f acc[4][2] = {};

    // fallback staging map: thread -> (row = tid>>3, 32 elems at (tid&7)*32)
    const int srow = tid >> 3;
    const int scol = (tid & 7) * 32;
    const unsigned short* wsrc = W + (long)(n0 + srow) * K + scol;
    unsigned short*       sdst = &Bs[srow * BSTRIDE + scol];
    (void)wsrc; (void)sdst;

    for (int kc = 0; kc < K; kc += KCHUNK) {
        __syncthreads();
#if HAVE_TDM
        // One wave DMAs the whole 32 x KCHUNK tile; HW inserts the 4-DWORD
        // row pad (pad_interval code 6 = 128 DWORDs, pad_amount code 3 = 4).
        if (wave == 0) {
            tdm_load_tile_2d(W + (long)n0 * K + kc,
                             (unsigned)(uintptr_t)&Bs[0],
                             K, KCHUNK, 32, 6u, 3u);
            __builtin_amdgcn_s_wait_tensorcnt(0);
        }
#elif HAVE_ASYNC_LDS
        #pragma unroll
        for (int q = 0; q < 4; ++q)
            async_copy_b128(wsrc + kc + q * 8, sdst + q * 8);
        __builtin_amdgcn_s_wait_asynccnt(0);
#else
        #pragma unroll
        for (int q = 0; q < 4; ++q)
            *(uint4*)(sdst + q * 8) = *(const uint4*)(wsrc + kc + q * 8);
#endif
        __syncthreads();

        if (kc + KCHUNK < K) {
            #pragma unroll
            for (int f = 0; f < 4; ++f)
                __builtin_prefetch(A + (long)(m0 + f * 16 + r) * K + kc + KCHUNK, 0, 1);
        }

        for (int ks = 0; ks < KCHUNK; ks += 32) {
            // B fragments from LDS (lane r = column, two 8-elem chunks @ hi*8, +16)
            union { uint4 u[2]; v16bf v; } bf0, bf1;
            {
                const unsigned short* bp0 = &Bs[(0 * 16 + r) * BSTRIDE + ks + kbB];
                const unsigned short* bp1 = &Bs[(1 * 16 + r) * BSTRIDE + ks + kbB];
                bf0.u[0] = *(const uint4*)(bp0);
                bf0.u[1] = *(const uint4*)(bp0 + 16);
                bf1.u[0] = *(const uint4*)(bp1);
                bf1.u[1] = *(const uint4*)(bp1 + 16);
            }
            #pragma unroll
            for (int f = 0; f < 4; ++f) {
                const unsigned short* ap =
                    A + (long)(m0 + f * 16 + r) * K + kc + ks + kbA;
                union { uint4 u[2]; v16bf v; } af;
                af.u[0] = *(const uint4*)(ap);
                af.u[1] = *(const uint4*)(ap + 16);
                acc[f][0] = __builtin_amdgcn_wmma_f32_16x16x32_bf16(
                    false, af.v, false, bf0.v, (short)0, acc[f][0], false, false);
                acc[f][1] = __builtin_amdgcn_wmma_f32_16x16x32_bf16(
                    false, af.v, false, bf1.v, (short)0, acc[f][1], false, false);
            }
        }
    }

    // C/D layout: VGPR j -> lanes 0-15: M=j, N=lane ; lanes 16-31: M=8+j, N=lane-16
    #pragma unroll
    for (int f = 0; f < 4; ++f)
        #pragma unroll
        for (int g = 0; g < 2; ++g)
            #pragma unroll
            for (int j = 0; j < 8; ++j)
                C[(long)(m0 + f * 16 + hi * 8 + j) * N + n0 + g * 16 + r] =
                    acc[f][g][j];
}

// ---------------------------------------------------------------------------
// Depthwise causal conv (K=4) + bias + SiLU over the xBC slice of zxbcdt
// ---------------------------------------------------------------------------
__global__ __launch_bounds__(256) void conv_silu_kernel(
    const float* __restrict__ zxbcdt,
    const float* __restrict__ conv_w,
    const float* __restrict__ conv_b,
    float* __restrict__ xBC)
{
    const int c   = blockIdx.x * blockDim.x + threadIdx.x;
    const int row = blockIdx.y;                // b*L + l
    if (c >= CONV_DIM) return;
    const int l = row % SEQLEN;

    float acc = conv_b[c];
    #pragma unroll
    for (int j = 0; j < D_CONV; ++j) {
        const int ll = l - (D_CONV - 1) + j;
        if (ll >= 0) {
            const float v = zxbcdt[(long)(row - l + ll) * D_IN_PROJ + D_INNER + c];
            acc += v * conv_w[c * D_CONV + j];
        }
    }
    xBC[(long)row * CONV_DIM + c] = acc / (1.0f + __expf(-acc));   // SiLU
}

// ---------------------------------------------------------------------------
// dt = softplus(raw_dt + dt_bias)
// ---------------------------------------------------------------------------
__global__ __launch_bounds__(256) void dt_softplus_kernel(
    const float* __restrict__ zxbcdt,
    const float* __restrict__ dt_bias,
    float* __restrict__ dt)
{
    const int i = blockIdx.x * blockDim.x + threadIdx.x;
    if (i >= ROWS * NHEADS) return;
    const int h = i & (NHEADS - 1);
    const float v = zxbcdt[(long)(i >> 5) * D_IN_PROJ + D_INNER + CONV_DIM + h]
                  + dt_bias[h];
    dt[i] = (v > 20.0f) ? v : __logf(1.0f + __expf(v));
}

// ---------------------------------------------------------------------------
// SSD scan: one block per (batch, head). State h[64][128] in registers
// (32 f32 per thread). B/C vectors staged in LDS each timestep (async path
// when available: 64 lanes x b128 = 1KB burst).
// thread t: p = t>>2, owns n in [ (t&3)*32, (t&3)*32+32 )
// ---------------------------------------------------------------------------
__global__ __launch_bounds__(256) void ssd_scan_kernel(
    const float* __restrict__ xBC,     // [ROWS, CONV_DIM]
    const float* __restrict__ dt,      // [ROWS, NHEADS]
    const float* __restrict__ A_log,
    const float* __restrict__ D_skip,
    float* __restrict__ y)             // [ROWS, D_INNER]
{
    const int b = blockIdx.x / NHEADS;
    const int h = blockIdx.x % NHEADS;
    const int t = threadIdx.x;
    const int p     = t >> 2;
    const int quad  = t & 3;
    const int nbase = quad * 32;

    const float Ah = -__expf(A_log[h]);
    const float Dh = D_skip[h];

    __shared__ float BCs[2 * D_STATE];   // [0,128): B ; [128,256): C

    float hreg[32];
    #pragma unroll
    for (int i = 0; i < 32; ++i) hreg[i] = 0.0f;

    for (int l = 0; l < SEQLEN; ++l) {
        const long row = (long)b * SEQLEN + l;
        __syncthreads();
#if HAVE_ASYNC_LDS
        if (t < 64)
            async_copy_b128(xBC + row * CONV_DIM + D_INNER + t * 4, &BCs[t * 4]);
        __builtin_amdgcn_s_wait_asynccnt(0);
#else
        if (t < 2 * D_STATE)
            BCs[t] = xBC[row * CONV_DIM + D_INNER + t];
#endif
        __syncthreads();

        const float dt_t = dt[row * NHEADS + h];
        const float dA   = __expf(dt_t * Ah);
        const float x_t  = xBC[row * CONV_DIM + h * HEADDIM + p];
        const float coef = dt_t * x_t;

        const float* Bsv = BCs;
        const float* Csv = BCs + D_STATE;

        float acc = 0.0f;
        #pragma unroll
        for (int i = 0; i < 32; ++i) {
            hreg[i] = dA * hreg[i] + coef * Bsv[nbase + i];
            acc    += hreg[i] * Csv[nbase + i];
        }
        acc += __shfl_xor(acc, 1, 32);
        acc += __shfl_xor(acc, 2, 32);
        if (quad == 0)
            y[row * D_INNER + h * HEADDIM + p] = acc + Dh * x_t;
    }
}

// ---------------------------------------------------------------------------
// y = y * silu(z); RMSNorm over D_INNER with norm_w; emit bf16 for out-proj
// ---------------------------------------------------------------------------
__global__ __launch_bounds__(256) void gate_rmsnorm_kernel(
    const float* __restrict__ y,
    const float* __restrict__ zxbcdt,
    const float* __restrict__ norm_w,
    unsigned short* __restrict__ y_bf16)
{
    const long row = blockIdx.x;
    const int tid  = threadIdx.x;
    const int lane = tid & 31;
    const int wave = tid >> 5;
    __shared__ float red[8];

    float vals[8];
    float ss = 0.0f;
    #pragma unroll
    for (int i = 0; i < 8; ++i) {
        const int c = tid + i * 256;
        const float z = zxbcdt[row * D_IN_PROJ + c];
        const float v = y[row * D_INNER + c] * (z / (1.0f + __expf(-z)));
        vals[i] = v;
        ss += v * v;
    }
    #pragma unroll
    for (int o = 16; o > 0; o >>= 1) ss += __shfl_xor(ss, o, 32);
    if (lane == 0) red[wave] = ss;
    __syncthreads();
    float total = 0.0f;
    #pragma unroll
    for (int w = 0; w < 8; ++w) total += red[w];
    const float rstd = rsqrtf(total / (float)D_INNER + EPS_F);

    #pragma unroll
    for (int i = 0; i < 8; ++i) {
        const int c = tid + i * 256;
        y_bf16[row * D_INNER + c] = f2bf(vals[i] * rstd * norm_w[c]);
    }
}

// ---------------------------------------------------------------------------
// LayerNorm over D_MODEL + residual add -> final output (fp32)
// ---------------------------------------------------------------------------
__global__ __launch_bounds__(256) void layernorm_residual_kernel(
    const float* __restrict__ out_lin,
    const float* __restrict__ x,
    const float* __restrict__ ln_g,
    const float* __restrict__ ln_b,
    float* __restrict__ out)
{
    const long row = blockIdx.x;
    const int tid  = threadIdx.x;
    const int lane = tid & 31;
    const int wave = tid >> 5;
    __shared__ float redS[8];
    __shared__ float redS2[8];

    float v[4];
    float s = 0.0f, s2 = 0.0f;
    #pragma unroll
    for (int i = 0; i < 4; ++i) {
        const int c = tid + i * 256;
        const float t = out_lin[row * D_MODEL + c];
        v[i] = t; s += t; s2 += t * t;
    }
    #pragma unroll
    for (int o = 16; o > 0; o >>= 1) {
        s  += __shfl_xor(s,  o, 32);
        s2 += __shfl_xor(s2, o, 32);
    }
    if (lane == 0) { redS[wave] = s; redS2[wave] = s2; }
    __syncthreads();
    float ts = 0.0f, ts2 = 0.0f;
    #pragma unroll
    for (int w = 0; w < 8; ++w) { ts += redS[w]; ts2 += redS2[w]; }
    const float mu   = ts / (float)D_MODEL;
    const float var  = ts2 / (float)D_MODEL - mu * mu;
    const float rstd = rsqrtf(var + EPS_F);

    #pragma unroll
    for (int i = 0; i < 4; ++i) {
        const int c = tid + i * 256;
        out[row * D_MODEL + c] =
            (v[i] - mu) * rstd * ln_g[c] + ln_b[c] + x[row * D_MODEL + c];
    }
}

// ---------------------------------------------------------------------------
// Launch
// ---------------------------------------------------------------------------
extern "C" void kernel_launch(void* const* d_in, const int* in_sizes, int n_in,
                              void* d_out, int out_size, void* d_ws, size_t ws_size,
                              hipStream_t stream) {
    const float* x         = (const float*)d_in[0];
    const float* in_proj_w = (const float*)d_in[1];
    const float* conv_w    = (const float*)d_in[2];
    const float* conv_b    = (const float*)d_in[3];
    const float* dt_bias   = (const float*)d_in[4];
    const float* A_log     = (const float*)d_in[5];
    const float* D_skip    = (const float*)d_in[6];
    const float* norm_w    = (const float*)d_in[7];
    const float* out_w     = (const float*)d_in[8];
    const float* ln_g      = (const float*)d_in[9];
    const float* ln_b      = (const float*)d_in[10];
    float* out = (float*)d_out;

    char* ws = (char*)d_ws;
    size_t off = 0;
    auto alloc = [&](size_t bytes) -> void* {
        void* p = ws + off;
        off = (off + bytes + 255) & ~(size_t)255;
        return p;
    };

    unsigned short* x_bf     = (unsigned short*)alloc((size_t)ROWS * D_MODEL * 2);
    unsigned short* win_bf   = (unsigned short*)alloc((size_t)D_IN_PROJ * D_MODEL * 2);
    unsigned short* wout_bf  = (unsigned short*)alloc((size_t)D_MODEL * D_INNER * 2);
    float*          zxbcdt   = (float*)alloc((size_t)ROWS * D_IN_PROJ * 4);
    float*          xBC      = (float*)alloc((size_t)ROWS * CONV_DIM * 4);
    float*          dt_buf   = (float*)alloc((size_t)ROWS * NHEADS * 4);
    float*          y_buf    = (float*)alloc((size_t)ROWS * D_INNER * 4);
    unsigned short* y_bf     = (unsigned short*)alloc((size_t)ROWS * D_INNER * 2);
    float*          out_lin  = (float*)alloc((size_t)ROWS * D_MODEL * 4);

    // 1) casts to bf16
    {
        long n1 = (long)ROWS * D_MODEL;
        cast_f32_bf16_kernel<<<dim3((n1 + 255) / 256), dim3(256), 0, stream>>>(x, x_bf, n1);
        long n2 = (long)D_IN_PROJ * D_MODEL;
        cast_f32_bf16_kernel<<<dim3((n2 + 255) / 256), dim3(256), 0, stream>>>(in_proj_w, win_bf, n2);
        long n3 = (long)D_MODEL * D_INNER;
        cast_f32_bf16_kernel<<<dim3((n3 + 255) / 256), dim3(256), 0, stream>>>(out_w, wout_bf, n3);
    }

    // 2) in_proj GEMM: zxbcdt[4096,4384] = x[4096,1024] @ in_proj_w[4384,1024]^T
    {
        dim3 grid(D_IN_PROJ / 32, ROWS / 512);   // (137, 8)
        wmma_gemm_bf16_kernel<<<grid, dim3(256), 0, stream>>>(
            x_bf, win_bf, zxbcdt, ROWS, D_IN_PROJ, D_MODEL);
    }

    // 3) depthwise causal conv + SiLU
    conv_silu_kernel<<<dim3((CONV_DIM + 255) / 256, ROWS), dim3(256), 0, stream>>>(
        zxbcdt, conv_w, conv_b, xBC);

    // 4) dt = softplus(raw + bias)
    dt_softplus_kernel<<<dim3((ROWS * NHEADS + 255) / 256), dim3(256), 0, stream>>>(
        zxbcdt, dt_bias, dt_buf);

    // 5) SSD scan
    ssd_scan_kernel<<<dim3(BATCH * NHEADS), dim3(256), 0, stream>>>(
        xBC, dt_buf, A_log, D_skip, y_buf);

    // 6) gate + RMSNorm -> bf16
    gate_rmsnorm_kernel<<<dim3(ROWS), dim3(256), 0, stream>>>(
        y_buf, zxbcdt, norm_w, y_bf);

    // 7) out-proj GEMM: out_lin[4096,1024] = y[4096,2048] @ out_w[1024,2048]^T
    {
        dim3 grid(D_MODEL / 32, ROWS / 512);     // (32, 8)
        wmma_gemm_bf16_kernel<<<grid, dim3(256), 0, stream>>>(
            y_bf, wout_bf, out_lin, ROWS, D_MODEL, D_INNER);
    }

    // 8) LayerNorm + residual
    layernorm_residual_kernel<<<dim3(ROWS), dim3(256), 0, stream>>>(
        out_lin, x, ln_g, ln_b, out);
}
